// DConv_70549132804437
// MI455X (gfx1250) — matrix operand order repaired and checked by reading, not compiled, not run on hardware
//
#include <hip/hip_runtime.h>
#include <cstdint>
#include <cstddef>

// ---------------------------------------------------------------------------
// Problem constants (from reference)
// ---------------------------------------------------------------------------
#define BB   8
#define CC   256      // C1 == C == C2 == 256
#define HH   64
#define WW   64
#define HWSZ 4096     // H*W
#define NPOS 32768    // B*H*W
#define GG   16
#define CG   16
#define KK   9
#define NOFF 432      // G*3*K

// ---------------------------------------------------------------------------
// WMMA types / helpers (CDNA5 wave32, v_wmma_f32_16x16x32_bf16)
// ---------------------------------------------------------------------------
typedef __attribute__((ext_vector_type(16))) __bf16 v16bf;
typedef __attribute__((ext_vector_type(8)))  float  v8f;

union ABFrag { unsigned u[8]; v16bf v; };

__device__ __forceinline__ unsigned short f2bf(float f) {
    unsigned u = __builtin_bit_cast(unsigned, f);
    u += 0x7fffu + ((u >> 16) & 1u);          // round-to-nearest-even
    return (unsigned short)(u >> 16);
}

__device__ __forceinline__ v8f wmma_bf16(const ABFrag& a, const ABFrag& b, v8f c) {
    // 8 args: (neg_a, A, neg_b, B, c_mod, C, reuse_a, reuse_b)
    return __builtin_amdgcn_wmma_f32_16x16x32_bf16(
        false, a.v, false, b.v, (short)0, c, false, false);
}

// A-fragment 16-bit layout (16x32): VGPR j holds packed K-pair at
//   kbase = (j>>2)*16 + (j&3)*2 + half*8   (half = lane>=16), M = lane&15.
__device__ __forceinline__ int a_kbase(int j, int half) {
    return (j >> 2) * 16 + (j & 3) * 2 + half * 8;
}
// B-fragment 16-bit layout (32x16): VGPR j holds K-pair at (j>>2)*16 + (j&3)*2 + half*8, N = lane&15.
__device__ __forceinline__ int b_kbase(int j, int half) {
    return (j >> 2) * 16 + (j & 3) * 2 + half * 8;
}

// ---------------------------------------------------------------------------
// Kernel 0: f32 -> bf16 weight conversion (runs once per launch; tiny)
// ---------------------------------------------------------------------------
__global__ void k_cvt_bf16(const float* __restrict__ src,
                           unsigned short* __restrict__ dst, int n) {
    int i = blockIdx.x * blockDim.x + threadIdx.x;
    if (i < n) dst[i] = f2bf(src[i]);
}

// ---------------------------------------------------------------------------
// Kernel 1: fused  h1 = x*W1^T + b1   (channel-last f32 + bf16-in-LDS)
//                  om = h1*Woff^T + boff
// One workgroup = 16 spatial positions (same b,h; 16 consecutive w).
// 8 waves: GEMM1 -> each wave 2 N-tiles (covers N=256); GEMM2 -> 27 N-tiles.
// ---------------------------------------------------------------------------
__global__ __launch_bounds__(256)
void k_cv1_off(const float* __restrict__ x,
               const float* __restrict__ b_cv1,
               const float* __restrict__ b_off,
               const unsigned short* __restrict__ wb1,
               const unsigned short* __restrict__ wboff,
               float* __restrict__ h1t,      // [NPOS][256] f32 (sampling values)
               float* __restrict__ om)       // [NPOS][432] f32 (offsets+mask)
{
    __shared__ unsigned short ldsA[16 * 256];   // x tile, bf16 row-major [m][k]
    __shared__ unsigned short ldsH[16 * 256];   // h1 tile, bf16 row-major [m][k]

    const int posbase = blockIdx.x * 16;
    const int b   = posbase >> 12;
    const int hw0 = posbase & 4095;             // h*64 + w0 (w0 multiple of 16)
    const int tid = threadIdx.x;

    // ---- stage x tile (16 pos x 256 ch) into LDS as bf16 ----
    {
        const int c = tid;                      // one channel plane per thread
        const float* src = x + ((size_t)(b * CC + c)) * HWSZ + hw0;
        #pragma unroll
        for (int m = 0; m < 16; ++m)
            ldsA[m * 256 + c] = f2bf(src[m]);
    }
    __syncthreads();

    const int lane = tid & 31;
    const int wave = tid >> 5;
    const int n    = lane & 15;
    const int half = lane >> 4;
    const int mA   = lane & 15;                 // A-fragment row

    // ---- GEMM1: h1 = A(16x256) * W1^T, N tiles per wave ----
    #pragma unroll
    for (int t = 0; t < 2; ++t) {
        const int nbase = (wave * 2 + t) * 16;
        v8f acc = {};
        #pragma unroll
        for (int ks = 0; ks < 256; ks += 32) {
            ABFrag a, bm;
            #pragma unroll
            for (int j = 0; j < 8; ++j) {
                a.u[j]  = *(const unsigned*)&ldsA[mA * 256 + ks + a_kbase(j, half)];
                bm.u[j] = *(const unsigned*)&wb1[(size_t)(nbase + n) * 256 + ks + b_kbase(j, half)];
            }
            acc = wmma_bf16(a, bm, acc);
        }
        const float bias = b_cv1[nbase + n];
        #pragma unroll
        for (int r = 0; r < 8; ++r) {
            const int m = r + 8 * half;         // C/D layout
            const float v = acc[r] + bias;
            h1t[(size_t)(posbase + m) * 256 + nbase + n] = v;   // f32 for sampler
            ldsH[m * 256 + nbase + n] = f2bf(v);                // bf16 for GEMM2
        }
    }
    __syncthreads();

    // ---- GEMM2: om = h1(16x256) * Woff^T, N = 432 -> 27 tiles across 8 waves ----
    for (int nt = wave; nt < 27; nt += 8) {
        const int nbase = nt * 16;
        v8f acc = {};
        #pragma unroll
        for (int ks = 0; ks < 256; ks += 32) {
            ABFrag a, bm;
            #pragma unroll
            for (int j = 0; j < 8; ++j) {
                a.u[j]  = *(const unsigned*)&ldsH[mA * 256 + ks + a_kbase(j, half)];
                bm.u[j] = *(const unsigned*)&wboff[(size_t)(nbase + n) * 256 + ks + b_kbase(j, half)];
            }
            acc = wmma_bf16(a, bm, acc);
        }
        const float bias = b_off[nbase + n];
        #pragma unroll
        for (int r = 0; r < 8; ++r) {
            const int m = r + 8 * half;
            om[(size_t)(posbase + m) * NOFF + nbase + n] = acc[r] + bias;
        }
    }
}

// ---------------------------------------------------------------------------
// Kernel 2: deformable bilinear sampling (gather-bound; h1t fits in 192MB L2).
// One block per spatial position; thread = (g = tid>>4, c = tid&15).
// Writes y in channel-last bf16 for the cv2 WMMA A-matrix.
// ---------------------------------------------------------------------------
__global__ __launch_bounds__(256)
void k_sample(const float* __restrict__ h1t,
              const float* __restrict__ om,
              unsigned short* __restrict__ ybf)    // [NPOS][256] bf16
{
    const int pos = blockIdx.x;
    const int b = pos >> 12, hw = pos & 4095;
    const int h = hw >> 6,  w  = hw & 63;
    const int tid = threadIdx.x;
    const int g = tid >> 4;                        // c = tid & 15 folded into base

    const float* o    = om  + (size_t)pos * NOFF + g * 27;     // 18 off + 9 mask
    const float* base = h1t + (size_t)b * HWSZ * 256 + tid;    // g*16 + c == tid

    const float KHk[9] = {-1.f,-1.f,-1.f, 0.f,0.f,0.f, 1.f,1.f,1.f};
    const float KWk[9] = {-1.f, 0.f, 1.f,-1.f,0.f,1.f,-1.f,0.f,1.f};

    auto tap = [&](int hi, int wi) -> float {
        if (hi < 0 || hi >= HH || wi < 0 || wi >= WW) return 0.f;
        return base[(size_t)(hi * WW + wi) * 256];
    };

    float acc = 0.f;
    #pragma unroll
    for (int k = 0; k < 9; ++k) {
        const float ph = (float)h + KHk[k] + o[2 * k];
        const float pw = (float)w + KWk[k] + o[2 * k + 1];
        const float h0f = floorf(ph), w0f = floorf(pw);
        const int   h0 = (int)h0f,    w0 = (int)w0f;
        const float ah = ph - h0f,    aw = pw - w0f;
        const float s = (1.f - ah) * (1.f - aw) * tap(h0,     w0)
                      + (1.f - ah) *        aw  * tap(h0,     w0 + 1)
                      +        ah  * (1.f - aw) * tap(h0 + 1, w0)
                      +        ah  *        aw  * tap(h0 + 1, w0 + 1);
        acc += o[18 + k] * s;
    }
    ybf[(size_t)pos * 256 + tid] = f2bf(acc);
}

// ---------------------------------------------------------------------------
// Kernel 3: cv2 GEMM (bf16 WMMA) + BatchNorm + SiLU, output in (B,C2,H,W) f32.
// ---------------------------------------------------------------------------
__global__ __launch_bounds__(256)
void k_cv2_bn(const unsigned short* __restrict__ ybf,
              const unsigned short* __restrict__ wb2,
              const float* __restrict__ bn_gamma,
              const float* __restrict__ bn_beta,
              const float* __restrict__ bn_mean,
              const float* __restrict__ bn_var,
              float* __restrict__ out)
{
    const int posbase = blockIdx.x * 16;
    const int b   = posbase >> 12;
    const int hw0 = posbase & 4095;
    const int tid  = threadIdx.x;
    const int lane = tid & 31;
    const int wave = tid >> 5;
    const int n    = lane & 15;
    const int half = lane >> 4;
    const int mA   = lane & 15;

    #pragma unroll
    for (int t = 0; t < 2; ++t) {
        const int nbase = (wave * 2 + t) * 16;
        const int ng    = nbase + n;
        v8f acc = {};
        #pragma unroll
        for (int ks = 0; ks < 256; ks += 32) {
            ABFrag a, bm;
            #pragma unroll
            for (int j = 0; j < 8; ++j) {
                a.u[j]  = *(const unsigned*)&ybf[(size_t)(posbase + mA) * 256 + ks + a_kbase(j, half)];
                bm.u[j] = *(const unsigned*)&wb2[(size_t)ng * 256 + ks + b_kbase(j, half)];
            }
            acc = wmma_bf16(a, bm, acc);
        }
        const float sc = bn_gamma[ng] * __frsqrt_rn(bn_var[ng] + 1e-5f);
        const float sh = bn_beta[ng] - bn_mean[ng] * sc;
        float* op = out + ((size_t)(b * CC + ng)) * HWSZ + hw0;
        #pragma unroll
        for (int r = 0; r < 8; ++r) {
            const int m = r + 8 * half;
            const float z = acc[r] * sc + sh;
            op[m] = z / (1.f + __expf(-z));     // z * sigmoid(z)
        }
    }
}

// ---------------------------------------------------------------------------
// Host launcher
// ---------------------------------------------------------------------------
extern "C" void kernel_launch(void* const* d_in, const int* in_sizes, int n_in,
                              void* d_out, int out_size, void* d_ws, size_t ws_size,
                              hipStream_t stream) {
    (void)in_sizes; (void)n_in; (void)out_size; (void)ws_size;

    const float* x        = (const float*)d_in[0];
    const float* w_cv1    = (const float*)d_in[1];
    const float* b_cv1    = (const float*)d_in[2];
    const float* w_off    = (const float*)d_in[3];
    const float* b_off    = (const float*)d_in[4];
    const float* w_cv2    = (const float*)d_in[5];
    const float* bn_gamma = (const float*)d_in[6];
    const float* bn_beta  = (const float*)d_in[7];
    const float* bn_mean  = (const float*)d_in[8];
    const float* bn_var   = (const float*)d_in[9];

    // Workspace carve-up (all sections keep 4-byte alignment)
    char* w = (char*)d_ws;
    float* h1t = (float*)w;                 w += (size_t)NPOS * 256 * sizeof(float);
    float* om  = (float*)w;                 w += (size_t)NPOS * NOFF * sizeof(float);
    unsigned short* ybf   = (unsigned short*)w;  w += (size_t)NPOS * 256 * 2;
    unsigned short* wb1   = (unsigned short*)w;  w += (size_t)CC * CC * 2;
    unsigned short* wboff = (unsigned short*)w;  w += (size_t)NOFF * CC * 2;
    unsigned short* wb2   = (unsigned short*)w;

    // Weight conversions (f32 -> bf16)
    k_cvt_bf16<<<(CC * CC + 255) / 256, 256, 0, stream>>>(w_cv1, wb1, CC * CC);
    k_cvt_bf16<<<(NOFF * CC + 255) / 256, 256, 0, stream>>>(w_off, wboff, NOFF * CC);
    k_cvt_bf16<<<(CC * CC + 255) / 256, 256, 0, stream>>>(w_cv2, wb2, CC * CC);

    // Fused cv1 + offset GEMMs (WMMA)
    k_cv1_off<<<NPOS / 16, 256, 0, stream>>>(x, b_cv1, b_off, wb1, wboff, h1t, om);

    // Deformable bilinear sampling
    k_sample<<<NPOS, 256, 0, stream>>>(h1t, om, ybf);

    // cv2 GEMM (WMMA) + BN + SiLU
    k_cv2_bn<<<NPOS / 16, 256, 0, stream>>>(ybf, wb2, bn_gamma, bn_beta,
                                            bn_mean, bn_var, (float*)d_out);
}
